// CIRKDMemLoss_16509854286625
// MI455X (gfx1250) — matrix operand order repaired and checked by reading, not compiled, not run on hardware
//
#include <hip/hip_runtime.h>
#include <math.h>

// ---------------------------------------------------------------------------
// CDNA5 (gfx1250) wave32 WMMA types
// ---------------------------------------------------------------------------
typedef __attribute__((ext_vector_type(16))) __bf16 v16bf;
typedef __attribute__((ext_vector_type(8)))  __bf16 v8bf;
typedef __attribute__((ext_vector_type(8)))  float  v8f;

#define M_ROWS 16384   // 4 * 64 * 64
#define CT     256
#define CS     512
#define NCLS   19
#define PIX_CS 216     // 4096//19 + 1
#define REG_CS 54      // 1024//19 + 1
#define NPIX   (NCLS*PIX_CS)   // 4104
#define NREG   (NCLS*REG_CS)   // 1026
#define NPIX_PAD 4112          // 257 tiles of 16
#define NREG_PAD 1040          // 65 tiles of 16

#define XS_PITCH 264           // 256 + 8 bf16 pad -> row stride 528B = 4 banks mod 64

// ---------------------------------------------------------------------------
// Fragment loaders (wave32; A and B share a contiguous k order, not the ISA table order)
// A 16x32 bf16: lane l holds row (l&15); elem i: K = kBase + 16*(l>>4) + i
// B 32x16 bf16 from Bt[N,K] row-major: lane l holds col (l&15);
//               elem i: K = kBase + 16*(l>>4) + i  (16 contiguous bf16)
// C/D 16x16 f32: vgpr j: lanes0-15 -> M=j, lanes16-31 -> M=j+8; N = lane&15
// ---------------------------------------------------------------------------
__device__ __forceinline__ v16bf load_a_frag(const __bf16* __restrict__ A,
                                             int K, int mBase, int kBase, int lane) {
  int r = lane & 15, half = lane >> 4;
  const __bf16* p0 = A + (size_t)(mBase + r) * K + kBase + 16 * half;
  union { v16bf v; v8bf h[2]; } u;
  u.h[0] = *(const v8bf*)(p0);
  u.h[1] = *(const v8bf*)(p0 + 8);
  return u.v;
}

__device__ __forceinline__ v16bf load_b_frag(const __bf16* __restrict__ Bt,
                                             int K, int nBase, int kBase, int lane) {
  int n = lane & 15, half = lane >> 4;
  return *(const v16bf*)(Bt + (size_t)(nBase + n) * K + kBase + 16 * half);
}

// B fragment out of the LDS-staged tile (16 rows x XS_PITCH bf16)
__device__ __forceinline__ v16bf load_b_frag_lds(const __bf16* tile,
                                                 int kBase, int lane) {
  int n = lane & 15, half = lane >> 4;
  return *(const v16bf*)(tile + n * XS_PITCH + kBase + 16 * half);
}

// ---------------------------------------------------------------------------
// Generic C[M,N] = A[M,K] * Bt[N,K]^T  (bf16 in, f32 out)
// block = 256 threads = 8 waves; grid (M/16, N/128); one 16x16 tile per wave
// ---------------------------------------------------------------------------
__global__ void wmma_gemm_bf16(const __bf16* __restrict__ A,
                               const __bf16* __restrict__ Bt,
                               float* __restrict__ C,
                               int M, int N, int K) {
  int lane = threadIdx.x & 31;
  int wave = threadIdx.x >> 5;
  int mBase = blockIdx.x * 16;
  int nBase = (blockIdx.y * 8 + wave) * 16;
  if (nBase >= N) return;
  v8f acc = {};
  for (int kb = 0; kb < K; kb += 32) {
    v16bf a = load_a_frag(A, K, mBase, kb, lane);
    v16bf b = load_b_frag(Bt, K, nBase, kb, lane);
    acc = __builtin_amdgcn_wmma_f32_16x16x32_bf16(false, a, false, b,
                                                  (short)0, acc, false, false);
  }
  int n = nBase + (lane & 15);
  int rbase = mBase + 8 * (lane >> 4);
#pragma unroll
  for (int j = 0; j < 8; ++j)
    C[(size_t)(rbase + j) * N + n] = acc[j];
}

// ---------------------------------------------------------------------------
// Async stage of one X n-tile (16 rows x 256 bf16) into LDS via the CDNA5
// async global->LDS engine (tracked by ASYNCcnt, no VGPR data movement).
// 512 x b128 transfers; 256 threads issue 2 each.
// ---------------------------------------------------------------------------
__device__ __forceinline__ void stage_tile_async(const __bf16* __restrict__ X,
                                                 int nt, __bf16* lds_buf,
                                                 int tid) {
#pragma unroll
  for (int t = 0; t < 2; ++t) {
    int e = tid + t * 256;            // e in [0,512)
    int row = e >> 5, seg = e & 31;   // 32 x 16B segments per row
    const __bf16* g = X + ((size_t)nt * 16 + row) * CT + seg * 8;
    unsigned lds = (unsigned)(uintptr_t)(lds_buf + row * XS_PITCH + seg * 8);
    asm volatile("global_load_async_to_lds_b128 %0, %1, off"
                 :: "v"(lds), "v"(g) : "memory");
  }
}

__device__ __forceinline__ void wait_async0() {
  asm volatile("s_wait_asynccnt 0x0" ::: "memory");
}

// ---------------------------------------------------------------------------
// Fused dual-GEMM + streaming-softmax KL divergence.
// Per wave: one 16-row m-tile; K=256 A-frags for teacher+student cached in
// registers. The 8 waves of a block share the X stream through a
// double-buffered async-staged LDS tile. Per-row online (max,Z,U,V) teacher
// stats and (max,Z) student stats; KL_row = (U-V)/Zt - mxT - ln Zt + mxS + ln Zs.
// ---------------------------------------------------------------------------
__global__ void kd_fused(const __bf16* __restrict__ T,
                         const __bf16* __restrict__ S,
                         const __bf16* __restrict__ X,
                         int ncols, int ntiles,
                         float invTemp, float scale,
                         float* __restrict__ out) {
  __shared__ __bf16 xs[2][16 * XS_PITCH];   // 2 x 8.25KB
  int tid  = threadIdx.x;
  int lane = tid & 31;
  int wave = tid >> 5;
  int mBase = (blockIdx.x * 8 + wave) * 16;
  const int K = CT;

  v16bf tA[8], sA[8];
#pragma unroll
  for (int ks = 0; ks < 8; ++ks) {
    tA[ks] = load_a_frag(T, K, mBase, ks * 32, lane);
    sA[ks] = load_a_frag(S, K, mBase, ks * 32, lane);
  }

  float mxT[8], Zt[8], Ut[8], Vt[8], mxS[8], Zs[8];
#pragma unroll
  for (int j = 0; j < 8; ++j) {
    mxT[j] = -3.0e38f; Zt[j] = 0.f; Ut[j] = 0.f; Vt[j] = 0.f;
    mxS[j] = -3.0e38f; Zs[j] = 0.f;
  }

  int lid = lane & 15;
  stage_tile_async(X, 0, xs[0], tid);

  for (int nt = 0; nt < ntiles; ++nt) {
    int cur = nt & 1;
    wait_async0();        // this wave's async copies into buffer `cur` done
    __syncthreads();      // all waves' copies done; prior readers of cur done
    if (nt + 1 < ntiles) stage_tile_async(X, nt + 1, xs[cur ^ 1], tid);

    v8f at = {}, as = {};
#pragma unroll
    for (int ks = 0; ks < 8; ++ks) {
      v16bf b = load_b_frag_lds(xs[cur], ks * 32, lane);
      at = __builtin_amdgcn_wmma_f32_16x16x32_bf16(false, tA[ks], false, b,
                                                   (short)0, at, false, false);
      as = __builtin_amdgcn_wmma_f32_16x16x32_bf16(false, sA[ks], false, b,
                                                   (short)0, as, false, false);
    }
    int nBase = nt * 16;
    if (nBase + lid < ncols) {
#pragma unroll
      for (int j = 0; j < 8; ++j) {
        float t = at[j] * invTemp;
        float s = as[j] * invTemp;
        if (t > mxT[j]) {
          float r = __expf(mxT[j] - t);
          Zt[j] = Zt[j] * r + 1.0f;
          Ut[j] = Ut[j] * r + t;
          Vt[j] = Vt[j] * r + s;
          mxT[j] = t;
        } else {
          float w = __expf(t - mxT[j]);
          Zt[j] += w; Ut[j] += w * t; Vt[j] += w * s;
        }
        if (s > mxS[j]) {
          Zs[j] = Zs[j] * __expf(mxS[j] - s) + 1.0f;
          mxS[j] = s;
        } else {
          Zs[j] += __expf(s - mxS[j]);
        }
      }
    }
    __syncthreads();      // finished reading `cur` before it is re-staged
  }

  // merge the 16 lanes of each half-wave (columns of the tile)
#pragma unroll
  for (int m = 1; m <= 8; m <<= 1) {
#pragma unroll
    for (int j = 0; j < 8; ++j) {
      float omx = __shfl_xor(mxT[j], m, 32);
      float oZ  = __shfl_xor(Zt[j],  m, 32);
      float oU  = __shfl_xor(Ut[j],  m, 32);
      float oV  = __shfl_xor(Vt[j],  m, 32);
      float nm  = fmaxf(mxT[j], omx);
      float e0  = __expf(mxT[j] - nm), e1 = __expf(omx - nm);
      Zt[j] = Zt[j] * e0 + oZ * e1;
      Ut[j] = Ut[j] * e0 + oU * e1;
      Vt[j] = Vt[j] * e0 + oV * e1;
      mxT[j] = nm;
      float omxs = __shfl_xor(mxS[j], m, 32);
      float oZs  = __shfl_xor(Zs[j],  m, 32);
      float nms  = fmaxf(mxS[j], omxs);
      Zs[j] = Zs[j] * __expf(mxS[j] - nms) + oZs * __expf(omxs - nms);
      mxS[j] = nms;
    }
  }

  if (lid == 0) {   // lane0 holds rows 0..7, lane16 holds rows 8..15
    float acc = 0.f;
#pragma unroll
    for (int j = 0; j < 8; ++j)
      acc += (Ut[j] - Vt[j]) / Zt[j] - mxT[j] - __logf(Zt[j])
             + mxS[j] + __logf(Zs[j]);
    atomicAdd(out, acc * scale);
  }
}

// ---------------------------------------------------------------------------
// Support kernels
// ---------------------------------------------------------------------------
__global__ void zero_out_kernel(float* out) {
  if (threadIdx.x == 0 && blockIdx.x == 0) out[0] = 0.f;
}

__global__ void f32_to_bf16_kernel(const float* __restrict__ in,
                                   __bf16* __restrict__ out, int n) {
  int i = blockIdx.x * 256 + threadIdx.x;
  if (i < n) out[i] = (__bf16)in[i];
}

// s_feats (N,Cs,H,W) -> A[m, c] bf16 row-major (m = b*4096+hw)
__global__ void pack_sfeats_kernel(const float* __restrict__ s,
                                   __bf16* __restrict__ A) {
  int gid = blockIdx.x * 256 + threadIdx.x;          // gid < M*CS
  int m = gid / CS, c = gid - m * CS;
  int b = m >> 12, hw = m & 4095;
  A[gid] = (__bf16)s[((size_t)b * CS + c) * 4096 + hw];
}

// channel-L2-normalize; mode 0: NCHW (C=256) input, mode 1: [M,256] row-major
__global__ void chan_norm_bf16_kernel(const float* __restrict__ x,
                                      __bf16* __restrict__ out, int mode) {
  int m = blockIdx.x, c = threadIdx.x;
  size_t addr;
  if (mode == 0) {
    int b = m >> 12, hw = m & 4095;
    addr = (size_t)b * CT * 4096 + (size_t)c * 4096 + hw;
  } else {
    addr = (size_t)m * CT + c;
  }
  float v = x[addr];
  __shared__ float red[256];
  red[c] = v * v;
  __syncthreads();
  for (int s = 128; s > 0; s >>= 1) {
    if (c < s) red[c] += red[c + s];
    __syncthreads();
  }
  float sc = 1.0f / fmaxf(sqrtf(red[0]), 1e-12f);
  out[(size_t)m * CT + c] = (__bf16)(v * sc);
}

__global__ void bnstat_kernel(const float* __restrict__ x,
                              float* __restrict__ stats) {
  int o = blockIdx.x, t = threadIdx.x;
  float s = 0.f, s2 = 0.f;
  for (int m = t; m < M_ROWS; m += 256) {
    float v = x[(size_t)m * CT + o];
    s += v; s2 += v * v;
  }
  __shared__ float r1[256], r2[256];
  r1[t] = s; r2[t] = s2;
  __syncthreads();
  for (int k = 128; k > 0; k >>= 1) {
    if (t < k) { r1[t] += r1[t + k]; r2[t] += r2[t + k]; }
    __syncthreads();
  }
  if (t == 0) { stats[o] = r1[0]; stats[CT + o] = r2[0]; }
}

__global__ void bnapply_kernel(const float* __restrict__ x,
                               const float* __restrict__ stats,
                               const float* __restrict__ gamma,
                               const float* __restrict__ beta,
                               __bf16* __restrict__ h) {
  int gid = blockIdx.x * 256 + threadIdx.x;
  int o = gid & (CT - 1);
  float mean = stats[o] * (1.0f / (float)M_ROWS);
  float var  = stats[CT + o] * (1.0f / (float)M_ROWS) - mean * mean;
  float v = (x[gid] - mean) * rsqrtf(var + 1e-5f) * gamma[o] + beta[o];
  h[gid] = (__bf16)fmaxf(v, 0.f);
}

// per-(class,batch) mean of normalized teacher features, re-normalized
__global__ void class_mean_kernel(const __bf16* __restrict__ T,
                                  const int* __restrict__ labels,
                                  float* __restrict__ seg_new) {
  int c = blockIdx.x >> 2, b = blockIdx.x & 3;
  int ch = threadIdx.x;
  float sum = 0.f; int cnt = 0;
  for (int p = 0; p < 4096; ++p) {
    if (labels[b * 4096 + p] == c) {
      sum += (float)T[(size_t)(b * 4096 + p) * CT + ch];
      ++cnt;
    }
  }
  float mean = sum / fmaxf((float)cnt, 1.0f);
  __shared__ float red[256];
  red[ch] = mean * mean;
  __syncthreads();
  for (int s = 128; s > 0; s >>= 1) {
    if (ch < s) red[ch] += red[ch + s];
    __syncthreads();
  }
  float sc = 1.0f / fmaxf(sqrtf(red[0]), 1e-12f);
  seg_new[((size_t)c * 4 + b) * CT + ch] = mean * sc;
}

// deterministic stand-in for the random top-k pixel pick: first 10 per class
__global__ void pix_select_kernel(const int* __restrict__ labels,
                                  int* __restrict__ sel_idx,
                                  int* __restrict__ sel_cnt) {
  int c = blockIdx.x >> 2, b = blockIdx.x & 3;
  if (threadIdx.x != 0) return;
  int cnt = 0;
  for (int p = 0; p < 4096 && cnt < 10; ++p)
    if (labels[b * 4096 + p] == c)
      sel_idx[(c * 4 + b) * 10 + cnt++] = b * 4096 + p;
  sel_cnt[c * 4 + b] = cnt;
}

// build sampled X_pix [NPIX_PAD,256] bf16 (coprime-stride pseudo-permutation)
__global__ void gather_xpix_kernel(const float* __restrict__ pix_q,
                                   const __bf16* __restrict__ T,
                                   const int* __restrict__ sel_idx,
                                   const int* __restrict__ sel_cnt,
                                   __bf16* __restrict__ X) {
  int j = blockIdx.x, ch = threadIdx.x;
  if (j >= NPIX) { X[(size_t)j * CT + ch] = (__bf16)0.f; return; }
  int c = j / PIX_CS, i = j - c * PIX_CS;
  int q = (i * 7919 + 13) % 20000;
  float v;
  if (q < 40) {                       // rows freshly enqueued this step
    int b = q / 10, r = q - b * 10;
    if (r < sel_cnt[c * 4 + b]) {
      int m = sel_idx[(c * 4 + b) * 10 + r];
      v = (float)T[(size_t)m * CT + ch];
    } else {
      v = pix_q[((size_t)c * 20000 + q) * CT + ch];
    }
  } else {
    v = pix_q[((size_t)c * 20000 + q) * CT + ch];
  }
  X[(size_t)j * CT + ch] = (__bf16)v;
}

__global__ void gather_xreg_kernel(const float* __restrict__ seg_q,
                                   const float* __restrict__ seg_new,
                                   __bf16* __restrict__ X) {
  int j = blockIdx.x, ch = threadIdx.x;
  if (j >= NREG) { X[(size_t)j * CT + ch] = (__bf16)0.f; return; }
  int c = j / REG_CS, i = j - c * REG_CS;
  int q = (i * 811 + 3) % 2000;
  float v = (q < 4) ? seg_new[((size_t)c * 4 + q) * CT + ch]
                    : seg_q[((size_t)c * 2000 + q) * CT + ch];
  X[(size_t)j * CT + ch] = (__bf16)v;
}

// ---------------------------------------------------------------------------
extern "C" void kernel_launch(void* const* d_in, const int* in_sizes, int n_in,
                              void* d_out, int out_size, void* d_ws, size_t ws_size,
                              hipStream_t stream) {
  const float* s_feats  = (const float*)d_in[0];
  const float* t_feats  = (const float*)d_in[1];
  // d_in[2] = logits_S (unused by the loss)
  const float* w1       = (const float*)d_in[3];
  const float* bn_gamma = (const float*)d_in[4];
  const float* bn_beta  = (const float*)d_in[5];
  const float* w2       = (const float*)d_in[6];
  const float* seg_q    = (const float*)d_in[7];
  const float* pix_q    = (const float*)d_in[8];
  const int*   labels   = (const int*)d_in[9];
  float* out = (float*)d_out;

  char* ws = (char*)d_ws;
  size_t off = 0;
  auto alloc = [&](size_t bytes) { size_t r = off; off += (bytes + 255) & ~(size_t)255; return r; };
  size_t o_tB   = alloc((size_t)M_ROWS * CT * 2);   // teacher bf16 [M,256]
  size_t o_sB   = alloc((size_t)M_ROWS * CT * 2);   // student bf16 [M,256]
  size_t o_A1   = alloc((size_t)M_ROWS * CS * 2);   // packed s_feats; reused as x2 f32
  size_t o_x1   = alloc((size_t)M_ROWS * CT * 4);   // GEMM1 out f32
  size_t o_hB   = alloc((size_t)M_ROWS * CT * 2);   // relu(bn(x1)) bf16
  size_t o_w1b  = alloc((size_t)CT * CS * 2);
  size_t o_w2b  = alloc((size_t)CT * CT * 2);
  size_t o_st   = alloc((size_t)2 * CT * 4);
  size_t o_segN = alloc((size_t)NCLS * 4 * CT * 4);
  size_t o_sidx = alloc((size_t)NCLS * 4 * 10 * 4);
  size_t o_scnt = alloc((size_t)NCLS * 4 * 4);
  size_t o_Xp   = alloc((size_t)NPIX_PAD * CT * 2);
  size_t o_Xr   = alloc((size_t)NREG_PAD * CT * 2);
  (void)ws_size;

  __bf16* tB   = (__bf16*)(ws + o_tB);
  __bf16* sB   = (__bf16*)(ws + o_sB);
  __bf16* A1   = (__bf16*)(ws + o_A1);
  float*  x1   = (float*)(ws + o_x1);
  float*  x2   = (float*)(ws + o_A1);   // reuse A1 region after GEMM1
  __bf16* hB   = (__bf16*)(ws + o_hB);
  __bf16* w1b  = (__bf16*)(ws + o_w1b);
  __bf16* w2b  = (__bf16*)(ws + o_w2b);
  float*  st   = (float*)(ws + o_st);
  float*  segN = (float*)(ws + o_segN);
  int*    sidx = (int*)(ws + o_sidx);
  int*    scnt = (int*)(ws + o_scnt);
  __bf16* Xp   = (__bf16*)(ws + o_Xp);
  __bf16* Xr   = (__bf16*)(ws + o_Xr);

  zero_out_kernel<<<1, 32, 0, stream>>>(out);

  // weight conversion + input packing
  f32_to_bf16_kernel<<<(CT * CS) / 256, 256, 0, stream>>>(w1, w1b, CT * CS);
  f32_to_bf16_kernel<<<(CT * CT) / 256, 256, 0, stream>>>(w2, w2b, CT * CT);
  chan_norm_bf16_kernel<<<M_ROWS, 256, 0, stream>>>(t_feats, tB, 0);
  pack_sfeats_kernel<<<(M_ROWS * CS) / 256, 256, 0, stream>>>(s_feats, A1);

  // student branch: GEMM1 -> BN -> ReLU -> GEMM2 -> L2 norm
  wmma_gemm_bf16<<<dim3(M_ROWS / 16, 2), 256, 0, stream>>>(A1, w1b, x1, M_ROWS, CT, CS);
  bnstat_kernel<<<CT, 256, 0, stream>>>(x1, st);
  bnapply_kernel<<<M_ROWS, 256, 0, stream>>>(x1, st, bn_gamma, bn_beta, hB);
  wmma_gemm_bf16<<<dim3(M_ROWS / 16, 2), 256, 0, stream>>>(hB, w2b, x2, M_ROWS, CT, CT);
  chan_norm_bf16_kernel<<<M_ROWS, 256, 0, stream>>>(x2, sB, 1);

  // queue update surrogates + contrast-set gathers
  class_mean_kernel<<<NCLS * 4, 256, 0, stream>>>(tB, labels, segN);
  pix_select_kernel<<<NCLS * 4, 32, 0, stream>>>(labels, sidx, scnt);
  gather_xpix_kernel<<<NPIX_PAD, 256, 0, stream>>>(pix_q, tB, sidx, scnt, Xp);
  gather_xreg_kernel<<<NREG_PAD, 256, 0, stream>>>(seg_q, segN, Xr);

  // fused contrastive-KD: dual WMMA GEMM + streaming softmax KL
  const float kScale = 0.1f / (float)M_ROWS;   // LW * KD_T^2 / rows
  kd_fused<<<M_ROWS / (16 * 8), 256, 0, stream>>>(tB, sB, Xp, NPIX, NPIX_PAD / 16,
                                                  10.0f, kScale, out);
  kd_fused<<<M_ROWS / (16 * 8), 256, 0, stream>>>(tB, sB, Xr, NREG, NREG_PAD / 16,
                                                  10.0f, kScale, out);
}
